// SpatialVisionTransformer_39599598469868
// MI455X (gfx1250) — matrix operand order repaired and checked by reading, not compiled, not run on hardware
//
#include <hip/hip_runtime.h>
#include <math.h>

// ---------------- types ----------------
typedef __bf16 bf16_t;
typedef __attribute__((ext_vector_type(16))) __bf16 v16bf;
typedef __attribute__((ext_vector_type(8)))  float  v8f;

union FragU { uint4 u[2]; v16bf v; };

#define B_    2
#define C_    768
#define NH_   12
#define HEAD_ 64
#define NP_   4
#define HP_   32
#define N_    1024           // tokens per image
#define M_    2048           // B_*N_
#define SCALE_ 0.125f        // HEAD^-0.5
#define EPS_  1e-6f

static __device__ __forceinline__ bf16_t f2bf(float f) {
    unsigned u = __builtin_bit_cast(unsigned, f);
    unsigned r = u + 0x7FFFu + ((u >> 16) & 1u);   // RNE
    unsigned short h = (unsigned short)(r >> 16);
    return __builtin_bit_cast(bf16_t, h);
}

// ---------------- fp32 -> bf16 convert ----------------
__global__ void k_cvt(const float* __restrict__ s, bf16_t* __restrict__ d, long n) {
    long i = (long)blockIdx.x * blockDim.x + threadIdx.x;
    long stride = (long)gridDim.x * blockDim.x;
    for (; i < n; i += stride) d[i] = f2bf(s[i]);
}

// ---------------- im2col for patch embed ----------------
__global__ void k_im2col(const float* __restrict__ x, bf16_t* __restrict__ Pm) {
    long i = (long)blockIdx.x * 256 + threadIdx.x;   // over 2048*768
    if (i >= (long)M_ * C_) return;
    int kidx = (int)(i % C_);
    long tok = i / C_;
    int n = (int)(tok % N_); int b = (int)(tok / N_);
    int hp = n >> 5, wp = n & 31;
    int ic = kidx >> 8; int rest = kidx & 255;
    int py = rest >> 4, px = rest & 15;
    int yy = hp * 16 + py, xx = wp * 16 + px;
    Pm[i] = f2bf(x[(((long)b * 3 + ic) * 512 + yy) * 512 + xx]);
}

// ---------------- layernorm (fp32 in -> bf16 out) ----------------
__global__ __launch_bounds__(256) void k_layernorm(
    const float* __restrict__ X, const float* __restrict__ s, const float* __restrict__ b,
    bf16_t* __restrict__ H) {
    long row = blockIdx.x;
    const float* xr = X + row * C_;
    __shared__ float red[256];
    float sum = 0.f;
    for (int c = threadIdx.x; c < C_; c += 256) sum += xr[c];
    red[threadIdx.x] = sum; __syncthreads();
    for (int o = 128; o > 0; o >>= 1) { if (threadIdx.x < o) red[threadIdx.x] += red[threadIdx.x + o]; __syncthreads(); }
    float mu = red[0] * (1.0f / C_); __syncthreads();
    float vs = 0.f;
    for (int c = threadIdx.x; c < C_; c += 256) { float d = xr[c] - mu; vs += d * d; }
    red[threadIdx.x] = vs; __syncthreads();
    for (int o = 128; o > 0; o >>= 1) { if (threadIdx.x < o) red[threadIdx.x] += red[threadIdx.x + o]; __syncthreads(); }
    float inv = rsqrtf(red[0] * (1.0f / C_) + EPS_);
    for (int c = threadIdx.x; c < C_; c += 256)
        H[row * C_ + c] = f2bf((xr[c] - mu) * inv * s[c] + b[c]);
}

// ---------------- batched WMMA bf16 GEMM ----------------
// C[m,n] = sum_k A[m,k] * W[n,k]  (+bias)(+gelu)(+resid into Cf)(+bf16 copy)
// slab z: off = (z/zdiv)*S1 + (z%zdiv)*S2  (element offsets)
// wave tile: 32(M) x 64(N); K pipelined 2-deep in 32-element blocks
// NOTE: requires K % 32 == 0 (all call sites: 64, 768, 1024, 3072)
#define GF_BIAS 1
#define GF_GELU 2
#define GF_RESID 4
#define GF_BF16 8
__global__ __launch_bounds__(128) void k_gemm(
    const bf16_t* __restrict__ A, const bf16_t* __restrict__ W,
    const float* __restrict__ bias, float* __restrict__ Cf, bf16_t* __restrict__ Cb,
    int M, int N, int K, int lda, int ldw, int ldc,
    int zdiv, long aS1, long aS2, long wS1, long wS2, long cS1, long cS2,
    int flags) {
    int lane = threadIdx.x;            // 0..31
    int wid  = threadIdx.y;            // 0..3
    long z = blockIdx.z;
    long zq = z / zdiv, zr = z % zdiv;
    A += zq * aS1 + zr * aS2;
    W += zq * wS1 + zr * wS2;
    long coff = zq * cS1 + zr * cS2;

    int m0 = blockIdx.y * 128 + wid * 32;   // wave: 32 rows
    int n0 = blockIdx.x * 64;               // block/wave: 64 cols
    if (m0 >= M) return;

    int lo = lane & 15;
    int hi = lane >> 4;                // 0 or 1
    int aRow0 = m0 + lo;      if (aRow0 >= M) aRow0 = M - 1;
    int aRow1 = m0 + 16 + lo; if (aRow1 >= M) aRow1 = M - 1;
    const bf16_t* aBase0 = A + (long)aRow0 * lda;
    const bf16_t* aBase1 = A + (long)aRow1 * lda;

    const bf16_t* wBase[4];
#pragma unroll
    for (int s4 = 0; s4 < 4; ++s4) {
        int col = n0 + s4 * 16 + lo;
        int cc = col < N ? col : N - 1;
        wBase[s4] = W + (long)cc * ldw;
    }

    v8f acc[2][4] = {{v8f{}, v8f{}, v8f{}, v8f{}}, {v8f{}, v8f{}, v8f{}, v8f{}}};

    // fragment load for one K-block of 32
    auto loadFrag = [&](int kk, FragU& a0, FragU& a1, FragU wf[4]) {
        a0.u[0] = *(const uint4*)(aBase0 + kk + hi * 8);
        a0.u[1] = *(const uint4*)(aBase0 + kk + hi * 8 + 16);
        a1.u[0] = *(const uint4*)(aBase1 + kk + hi * 8);
        a1.u[1] = *(const uint4*)(aBase1 + kk + hi * 8 + 16);
#pragma unroll
        for (int s4 = 0; s4 < 4; ++s4) {
            const bf16_t* wp = wBase[s4] + kk + hi * 8;
            wf[s4].u[0] = *(const uint4*)(wp);
            wf[s4].u[1] = *(const uint4*)(wp + 16);
        }
    };
    auto domma = [&](FragU& a0, FragU& a1, FragU wf[4]) {
#pragma unroll
        for (int s4 = 0; s4 < 4; ++s4) {
            acc[0][s4] = __builtin_amdgcn_wmma_f32_16x16x32_bf16(
                false, a0.v, false, wf[s4].v, (short)0, acc[0][s4], false, false);
            acc[1][s4] = __builtin_amdgcn_wmma_f32_16x16x32_bf16(
                false, a1.v, false, wf[s4].v, (short)0, acc[1][s4], false, false);
        }
    };

    FragU a0A, a1A, wfA[4], a0B, a1B, wfB[4];
    loadFrag(0, a0A, a1A, wfA);
    for (int k = 0; k < K; k += 64) {
        if (k + 32 < K) loadFrag(k + 32, a0B, a1B, wfB);   // next block in flight
        if (k + 128 < K) {                                  // global_prefetch_b8 hint
            __builtin_prefetch(aBase0 + k + 128 + hi * 8, 0, 3);
            __builtin_prefetch(aBase1 + k + 128 + hi * 8, 0, 3);
        }
        domma(a0A, a1A, wfA);
        if (k + 64 < K) loadFrag(k + 64, a0A, a1A, wfA);
        if (k + 32 < K) domma(a0B, a1B, wfB);
    }

    // store: VGPR r -> row mh*16 + hi*8 + r, col n0 + s4*16 + lo
#pragma unroll
    for (int mh = 0; mh < 2; ++mh) {
        int rowB = m0 + mh * 16 + hi * 8;
#pragma unroll
        for (int s4 = 0; s4 < 4; ++s4) {
            int col = n0 + s4 * 16 + lo;
            if (col >= N) continue;
            float bv = (flags & GF_BIAS) ? bias[col] : 0.f;
#pragma unroll
            for (int r = 0; r < 8; ++r) {
                int row = rowB + r;
                if (row >= M) continue;
                float v = acc[mh][s4][r] + bv;
                if (flags & GF_GELU) v = 0.5f * v * (1.f + erff(v * 0.70710678118f));
                long idx = coff + (long)row * ldc + col;
                if (flags & GF_RESID) v += Cf[idx];
                Cf[idx] = v;
                if (flags & GF_BF16) Cb[idx] = f2bf(v);
            }
        }
    }
}

// ---------------- deformable sampled attention ----------------
// one wave per (b,n,h); lane owns dims d and d+32
__global__ __launch_bounds__(128) void k_deform(
    const float* __restrict__ QKV, const float* __restrict__ OFFS,
    float* __restrict__ AO, bf16_t* __restrict__ AOb) {
    int lane = threadIdx.x;
    long widx = (long)blockIdx.x * 4 + threadIdx.y;     // over B_*N_*NH_
    if (widx >= (long)B_ * N_ * NH_) return;
    int h = (int)(widx % NH_);
    long bn = widx / NH_;
    int n = (int)(bn % N_); int b = (int)(bn / N_);
    int hp = n >> 5, wp = n & 31;

    const float* qrow  = QKV + ((long)(b * N_ + n)) * (3 * C_) + h * HEAD_;
    const float* krow0 = QKV + ((long)b * N_) * (3 * C_) + C_ + h * HEAD_;
    const float* vrow0 = krow0 + C_;
    const float* of    = OFFS + ((long)(b * N_ + n)) * (NH_ * NP_ * 2) + h * (NP_ * 2);

    int d0 = lane, d1 = lane + 32;
    float q0 = qrow[d0], q1 = qrow[d1];

    float logit[NP_], sv0[NP_], sv1[NP_];
#pragma unroll
    for (int p = 0; p < NP_; ++p) {
        float xx = (float)wp + of[p * 2];
        float yy = (float)hp + of[p * 2 + 1];
        float x0f = floorf(xx), y0f = floorf(yy);
        float wx = xx - x0f, wy = yy - y0f;
        int x0 = (int)x0f, y0 = (int)y0f;
        float k0 = 0.f, k1 = 0.f, v0 = 0.f, v1 = 0.f;
#pragma unroll
        for (int dy = 0; dy < 2; ++dy) {
#pragma unroll
            for (int dx = 0; dx < 2; ++dx) {
                int xi = x0 + dx, yi = y0 + dy;
                float w = (dx ? wx : 1.f - wx) * (dy ? wy : 1.f - wy);
                if (xi >= 0 && xi < HP_ && yi >= 0 && yi < HP_) {
                    long t = (long)(yi * HP_ + xi) * (3 * C_);
                    k0 += w * krow0[t + d0]; k1 += w * krow0[t + d1];
                    v0 += w * vrow0[t + d0]; v1 += w * vrow0[t + d1];
                }
            }
        }
        logit[p] = q0 * k0 + q1 * k1;
        sv0[p] = v0; sv1[p] = v1;
    }
    // reduce logits over 32 lanes
#pragma unroll
    for (int p = 0; p < NP_; ++p) {
        float v = logit[p];
        for (int m = 16; m > 0; m >>= 1) v += __shfl_xor(v, m, 32);
        logit[p] = v * SCALE_;
    }
    float mx = fmaxf(fmaxf(logit[0], logit[1]), fmaxf(logit[2], logit[3]));
    float e[NP_], es = 0.f;
#pragma unroll
    for (int p = 0; p < NP_; ++p) { e[p] = expf(logit[p] - mx); es += e[p]; }
    float inv = 1.f / es;
    float o0 = 0.f, o1 = 0.f;
#pragma unroll
    for (int p = 0; p < NP_; ++p) { o0 += e[p] * inv * sv0[p]; o1 += e[p] * inv * sv1[p]; }
    long orow = ((long)(b * N_ + n)) * C_ + h * HEAD_;
    AO[orow + d0] = o0; AO[orow + d1] = o1;
    AOb[orow + d0] = f2bf(o0); AOb[orow + d1] = f2bf(o1);
}

// ---------------- row softmax (S fp32 -> P bf16), with scale ----------------
__global__ __launch_bounds__(256) void k_softmax(
    const float* __restrict__ S, bf16_t* __restrict__ P, int cols, float scale) {
    long row = blockIdx.x;
    const float* sr = S + row * cols;
    bf16_t* pr = P + row * cols;
    __shared__ float red[256];
    float mx = -3.4e38f;
    for (int c = threadIdx.x; c < cols; c += 256) mx = fmaxf(mx, sr[c] * scale);
    red[threadIdx.x] = mx; __syncthreads();
    for (int o = 128; o > 0; o >>= 1) { if (threadIdx.x < o) red[threadIdx.x] = fmaxf(red[threadIdx.x], red[threadIdx.x + o]); __syncthreads(); }
    mx = red[0]; __syncthreads();
    float sum = 0.f;
    for (int c = threadIdx.x; c < cols; c += 256) sum += expf(sr[c] * scale - mx);
    red[threadIdx.x] = sum; __syncthreads();
    for (int o = 128; o > 0; o >>= 1) { if (threadIdx.x < o) red[threadIdx.x] += red[threadIdx.x + o]; __syncthreads(); }
    float inv = 1.f / red[0];
    for (int c = threadIdx.x; c < cols; c += 256)
        pr[c] = f2bf(expf(sr[c] * scale - mx) * inv);
}

// ---------------- V transpose per head: Vt[bh][d][m] ----------------
__global__ void k_vt(const bf16_t* __restrict__ QKVb, bf16_t* __restrict__ Vt) {
    long i = (long)blockIdx.x * 256 + threadIdx.x;   // 24*64*1024
    if (i >= (long)B_ * NH_ * HEAD_ * N_) return;
    int m = (int)(i & (N_ - 1));
    long r = i >> 10;
    int d = (int)(r & (HEAD_ - 1));
    long bh = r >> 6;
    int b = (int)(bh / NH_), h = (int)(bh % NH_);
    Vt[i] = QKVb[((long)(b * N_ + m)) * (3 * C_) + 2 * C_ + h * HEAD_ + d];
}

// ---------------- final NCHW write-out ----------------
__global__ void k_out(const float* __restrict__ X, float* __restrict__ out) {
    long i = (long)blockIdx.x * 256 + threadIdx.x;   // 2*768*1024
    if (i >= (long)B_ * C_ * N_) return;
    int n = (int)(i % N_);
    long r = i / N_;
    int c = (int)(r % C_); int b = (int)(r / C_);
    out[i] = X[((long)(b * N_ + n)) * C_ + c];
}

// ---------------- host ----------------
extern "C" void kernel_launch(void* const* d_in, const int* in_sizes, int n_in,
                              void* d_out, int out_size, void* d_ws, size_t ws_size,
                              hipStream_t stream) {
    const float* x       = (const float*)d_in[0];
    const float* patch_w = (const float*)d_in[1];
    const float* patch_b = (const float*)d_in[2];
    const float* ln1_s   = (const float*)d_in[3];
    const float* ln1_b   = (const float*)d_in[4];
    const float* qkv_w   = (const float*)d_in[5];
    const float* offs_w  = (const float*)d_in[6];
    const float* offs_b  = (const float*)d_in[7];
    const float* proj_w  = (const float*)d_in[8];
    const float* proj_b  = (const float*)d_in[9];
    const float* ln2_s   = (const float*)d_in[10];
    const float* ln2_b   = (const float*)d_in[11];
    const float* fc1_w   = (const float*)d_in[12];
    const float* fc1_b   = (const float*)d_in[13];
    const float* fc2_w   = (const float*)d_in[14];
    const float* fc2_b   = (const float*)d_in[15];

    char* ws = (char*)d_ws;
    size_t off = 0;
    auto alloc = [&](size_t bytes) { size_t o = off; off += (bytes + 255) & ~(size_t)255; return o; };

    const long nPatchW = (long)C_ * C_;                 // 768*768 (3*16*16 = 768)
    const long nQkvW   = 6L * 3 * C_ * C_;
    const long nOffsW  = 6L * NH_ * NP_ * 2 * C_;
    const long nProjW  = 6L * C_ * C_;
    const long nFc1W   = 6L * 4 * C_ * C_;
    const long nFc2W   = 6L * C_ * 4 * C_;

    bf16_t* patch_wb = (bf16_t*)(ws + alloc(nPatchW * 2));
    bf16_t* qkv_wb   = (bf16_t*)(ws + alloc(nQkvW * 2));
    bf16_t* offs_wb  = (bf16_t*)(ws + alloc(nOffsW * 2));
    bf16_t* proj_wb  = (bf16_t*)(ws + alloc(nProjW * 2));
    bf16_t* fc1_wb   = (bf16_t*)(ws + alloc(nFc1W * 2));
    bf16_t* fc2_wb   = (bf16_t*)(ws + alloc(nFc2W * 2));

    float*  X    = (float*)(ws + alloc((long)M_ * C_ * 4));
    bf16_t* H    = (bf16_t*)(ws + alloc((long)M_ * C_ * 2));         // also im2col buffer
    float*  QKV  = (float*)(ws + alloc((long)M_ * 3 * C_ * 4));
    bf16_t* QKVb = (bf16_t*)(ws + alloc((long)M_ * 3 * C_ * 2));
    float*  OFFS = (float*)(ws + alloc((long)M_ * NH_ * NP_ * 2 * 4));
    float*  ATTO = (float*)(ws + alloc((long)M_ * C_ * 4));
    bf16_t* ATTOb= (bf16_t*)(ws + alloc((long)M_ * C_ * 2));
    // S (24 * 1024*1024 f32) aliases FC1 (2048*3072 f32): disjoint lifetimes
    size_t sBytes   = (size_t)B_ * NH_ * N_ * N_ * 4;
    size_t fc1Bytes = (size_t)M_ * 4 * C_ * 4;
    float*  S    = (float*)(ws + alloc(sBytes > fc1Bytes ? sBytes : fc1Bytes));
    float*  FC1  = S;
    size_t pBytes   = (size_t)B_ * NH_ * N_ * N_ * 2;
    size_t fc1bB    = (size_t)M_ * 4 * C_ * 2;
    bf16_t* Pb   = (bf16_t*)(ws + alloc(pBytes > fc1bB ? pBytes : fc1bB));
    bf16_t* FC1b = Pb;
    bf16_t* Vt   = (bf16_t*)(ws + alloc((long)B_ * NH_ * HEAD_ * N_ * 2));

    auto cvt = [&](const float* s, bf16_t* d, long n) {
        int gb = (int)((n + 255) / 256); if (gb > 8192) gb = 8192;
        k_cvt<<<gb, 256, 0, stream>>>(s, d, n);
    };
    auto gemm = [&](const bf16_t* A, const bf16_t* W, const float* bias,
                    float* Cf, bf16_t* Cb, int M, int N, int K,
                    int lda, int ldw, int ldc, int nz, int zdiv,
                    long aS1, long aS2, long wS1, long wS2, long cS1, long cS2, int flags) {
        dim3 g((N + 63) / 64, (M + 127) / 128, nz);
        k_gemm<<<g, dim3(32, 4, 1), 0, stream>>>(A, W, bias, Cf, Cb, M, N, K,
            lda, ldw, ldc, zdiv, aS1, aS2, wS1, wS2, cS1, cS2, flags);
    };

    // ---- weight conversion (deterministic every call) ----
    cvt(patch_w, patch_wb, nPatchW);
    cvt(qkv_w,   qkv_wb,   nQkvW);
    cvt(offs_w,  offs_wb,  nOffsW);
    cvt(proj_w,  proj_wb,  nProjW);
    cvt(fc1_w,   fc1_wb,   nFc1W);
    cvt(fc2_w,   fc2_wb,   nFc2W);

    // ---- patch embed: im2col into H, then GEMM -> X ----
    k_im2col<<<(int)(((long)M_ * C_ + 255) / 256), 256, 0, stream>>>(x, H);
    gemm(H, patch_wb, patch_b, X, nullptr, M_, C_, C_, C_, C_, C_,
         1, 1, 0, 0, 0, 0, 0, 0, GF_BIAS);

    for (int i = 0; i < 6; ++i) {
        bool sample = ((i + 1) % 3) != 0;
        // ln1
        k_layernorm<<<M_, 256, 0, stream>>>(X, ln1_s + (long)i * C_, ln1_b + (long)i * C_, H);
        // qkv: fp32 + bf16
        gemm(H, qkv_wb + (long)i * 3 * C_ * C_, nullptr, QKV, QKVb,
             M_, 3 * C_, C_, C_, C_, 3 * C_, 1, 1, 0, 0, 0, 0, 0, 0, GF_BF16);
        if (sample) {
            // offsets = q @ offs_w.T + offs_b
            gemm(QKVb, offs_wb + (long)i * NH_ * NP_ * 2 * C_, offs_b + (long)i * NH_ * NP_ * 2,
                 OFFS, nullptr, M_, NH_ * NP_ * 2, C_, 3 * C_, C_, NH_ * NP_ * 2,
                 1, 1, 0, 0, 0, 0, 0, 0, GF_BIAS);
            k_deform<<<(int)(((long)B_ * N_ * NH_ + 3) / 4), dim3(32, 4, 1), 0, stream>>>(
                QKV, OFFS, ATTO, ATTOb);
        } else {
            // S = Q K^T per (b,h) slab; z = b*12 + h
            gemm(QKVb, QKVb + C_, nullptr, S, nullptr,
                 N_, N_, HEAD_, 3 * C_, 3 * C_, N_, B_ * NH_, NH_,
                 (long)N_ * 3 * C_, HEAD_,
                 (long)N_ * 3 * C_, HEAD_,
                 (long)NH_ * N_ * N_, (long)N_ * N_, 0);
            k_softmax<<<B_ * NH_ * N_, 256, 0, stream>>>(S, Pb, N_, SCALE_);
            k_vt<<<(int)(((long)B_ * NH_ * HEAD_ * N_ + 255) / 256), 256, 0, stream>>>(QKVb, Vt);
            // out = P V ; C written strided into ATTO at (b*N) rows, h*64 cols
            gemm(Pb, Vt, nullptr, ATTO, ATTOb,
                 N_, HEAD_, N_, N_, N_, C_, B_ * NH_, NH_,
                 (long)NH_ * N_ * N_, (long)N_ * N_,
                 (long)NH_ * HEAD_ * N_, (long)HEAD_ * N_,
                 (long)N_ * C_, (long)HEAD_, GF_BF16);
        }
        // proj (+bias) with residual into X
        gemm(ATTOb, proj_wb + (long)i * C_ * C_, proj_b + (long)i * C_, X, nullptr,
             M_, C_, C_, C_, C_, C_, 1, 1, 0, 0, 0, 0, 0, 0, GF_BIAS | GF_RESID);
        // ln2
        k_layernorm<<<M_, 256, 0, stream>>>(X, ln2_s + (long)i * C_, ln2_b + (long)i * C_, H);
        // fc1 + gelu
        gemm(H, fc1_wb + (long)i * 4 * C_ * C_, fc1_b + (long)i * 4 * C_, FC1, FC1b,
             M_, 4 * C_, C_, C_, C_, 4 * C_, 1, 1, 0, 0, 0, 0, 0, 0,
             GF_BIAS | GF_GELU | GF_BF16);
        // fc2 with residual into X
        gemm(FC1b, fc2_wb + (long)i * C_ * 4 * C_, fc2_b + (long)i * C_, X, nullptr,
             M_, C_, 4 * C_, 4 * C_, 4 * C_, C_, 1, 1, 0, 0, 0, 0, 0, 0,
             GF_BIAS | GF_RESID);
    }

    // ---- final (B, C, Hp, Wp) write-out ----
    k_out<<<(int)(((long)B_ * C_ * N_ + 255) / 256), 256, 0, stream>>>(X, (float*)d_out);

    (void)in_sizes; (void)n_in; (void)out_size; (void)ws_size;
}
